// DecoderRNNTAtt_62534723829918
// MI455X (gfx1250) — compile-verified, matrix operand order checked by reading, not run on hardware
//
#include <hip/hip_runtime.h>
#include <hip/hip_bf16.h>
#include <math.h>

typedef _Float16 f16;
typedef __attribute__((ext_vector_type(16))) _Float16 v16h;
typedef __attribute__((ext_vector_type(8)))  _Float16 v8h;
typedef __attribute__((ext_vector_type(8)))  float    v8f;
typedef __attribute__((ext_vector_type(4)))  int      v4i;
typedef __attribute__((address_space(1))) v4i* gv4i_p;  // global
typedef __attribute__((address_space(3))) v4i* lv4i_p;  // LDS

#define B_   8
#define T_   256
#define U_   64
#define E_   512
#define OD_  500

#if __has_builtin(__builtin_amdgcn_global_load_async_to_lds_b128) && \
    __has_builtin(__builtin_amdgcn_s_wait_asynccnt)
#define HAS_ASYNC_LDS 1
#else
#define HAS_ASYNC_LDS 0
#endif

// ---- workspace byte offsets ----
#define OFF_HS_H    0u          // (B*T,512) f16        2 MB
#define OFF_PRE_H   2097152u    // (B*T,512) f16        2 MB
#define OFF_ENCJ    4194304u    // (B*T,512) fp32       4 MB
#define OFF_DECJ    8388608u    // (B*U,512) fp32       1 MB
#define OFF_HDEC_H  9437184u    // (B*U,512) f16        0.5 MB
#define OFF_EYS_H   9961472u    // (B*U,512) f16        0.5 MB
#define OFF_WAE_H   10485760u   // swizzled (512,512)
#define OFF_WLE_H   11010048u   // swizzled (512,512)
#define OFF_WLD_H   11534336u   // swizzled (512,512)
#define OFF_WAD_H   12058624u   // swizzled (512,512)
#define OFF_WOUT_H  12582912u   // swizzled (512,512) padded from (512,500)
#define OFF_W0_H    13107200u   // swizzled (1536,2048) = [W_ih0^T;W_hh0^T] 6 MB
#define OFF_W1_H    19398656u   // swizzled (1024,2048) = [W_ih1^T;W_hh1^T] 4 MB

// ------------- WMMA fragment helpers (16-bit 16x16x32 lane layout) -----------
__device__ __forceinline__ v16h frag_b_sw(const f16* __restrict__ B, int NT, int nt,
                                          int kk, int lane) {
  return *(const v16h*)(B + ((size_t)((kk * NT + nt) * 32 + lane) << 4));
}

__device__ __forceinline__ v16h combine16(v8h lo, v8h hi) {
  return __builtin_shufflevector(lo, hi, 0, 1, 2, 3, 4, 5, 6, 7,
                                 8, 9, 10, 11, 12, 13, 14, 15);
}

__device__ __forceinline__ v16h frag_a_vec(const f16* __restrict__ Am, int rowstride,
                                           int k0, int lane) {
  int m = lane & 15, kb = (lane >> 4) * 8;
  const f16* p = Am + m * rowstride + k0 + kb;
  return combine16(*(const v8h*)p, *(const v8h*)(p + 16));
}

__device__ __forceinline__ v8f wmma16(v16h a, v16h b, v8f c) {
  return __builtin_amdgcn_wmma_f32_16x16x32_f16(false, a, false, b, (short)0, c,
                                                false, false);
}

__device__ __forceinline__ float sigm(float x) { return 1.0f / (1.0f + expf(-x)); }

// ---------------- prep kernels ----------------
__global__ void k_cast(f16* dst, const float* src, int n) {
  int i = blockIdx.x * blockDim.x + threadIdx.x;
  if (i < n) dst[i] = (f16)src[i];
}

__global__ void k_emb(f16* dst, const float* emb, const int* ys) {  // (B*U,512)
  int i = blockIdx.x * 256 + threadIdx.x;                           // 262144
  int bu = i >> 9, e = i & 511;
  dst[i] = (f16)emb[ys[bu] * 512 + e];
}

__global__ void k_sw_direct(f16* dst, const float* src, int NT, int total) {
  int i = blockIdx.x * 256 + threadIdx.x;
  if (i >= total) return;
  int e = i & 15, lane = (i >> 4) & 31, tile = i >> 9;
  int nt = tile % NT, kk = tile / NT;
  int k = kk * 32 + (lane >> 4) * 8 + (e < 8 ? e : e + 8);
  int n = nt * 16 + (lane & 15);
  dst[i] = (f16)src[k * (NT * 16) + n];
}

__global__ void k_sw_wout(f16* dst, const float* src) {  // (512,500) -> padded 512
  int i = blockIdx.x * 256 + threadIdx.x;                // 262144
  int e = i & 15, lane = (i >> 4) & 31, tile = i >> 9;
  int nt = tile & 31, kk = tile >> 5;
  int k = kk * 32 + (lane >> 4) * 8 + (e < 8 ? e : e + 8);
  int n = nt * 16 + (lane & 15);
  dst[i] = (n < OD_) ? (f16)src[k * OD_ + n] : (f16)0.0f;
}

__global__ void k_sw_stack0(f16* dst, const float* Wih, const float* Whh) { // 1536x2048
  int i = blockIdx.x * 256 + threadIdx.x;
  if (i >= 1536 * 2048) return;
  int e = i & 15, lane = (i >> 4) & 31, tile = i >> 9;
  int nt = tile & 127, kk = tile >> 7;
  int k = kk * 32 + (lane >> 4) * 8 + (e < 8 ? e : e + 8);
  int n = nt * 16 + (lane & 15);
  dst[i] = (f16)((k < 1024) ? Wih[n * 1024 + k] : Whh[n * 512 + (k - 1024)]);
}

__global__ void k_sw_stack1(f16* dst, const float* Wih, const float* Whh) { // 1024x2048
  int i = blockIdx.x * 256 + threadIdx.x;
  if (i >= 1024 * 2048) return;
  int e = i & 15, lane = (i >> 4) & 31, tile = i >> 9;
  int nt = tile & 127, kk = tile >> 7;
  int k = kk * 32 + (lane >> 4) * 8 + (e < 8 ? e : e + 8);
  int n = nt * 16 + (lane & 15);
  dst[i] = (f16)((k < 512) ? Wih[n * 512 + k] : Whh[n * 512 + (k - 512)]);
}

// ---------------- encoder projections: pre_enc (tanh,f16) + encJ (fp32) ------
__global__ __launch_bounds__(256) void k_enc(const f16* __restrict__ hs_h,
                                             const f16* __restrict__ WaeSW,
                                             const f16* __restrict__ WleSW,
                                             const float* __restrict__ b_ae,
                                             const float* __restrict__ b_le,
                                             f16* __restrict__ pre_h,
                                             float* __restrict__ encJ) {
  int lane = threadIdx.x & 31, wv = threadIdx.x >> 5;
  int gid = blockIdx.x * 8 + wv;  // 8192 wave-tiles: 2 outputs x 128mt x 32nt
  int which = gid >> 12;
  int rem = gid & 4095;
  int mt = rem >> 5, nt = rem & 31;
  const f16* Bm = which ? WleSW : WaeSW;
  const f16* Arow = hs_h + (mt * 16) * 512;
  v8f acc = {};
  for (int kk = 0; kk < 16; ++kk) {
    v16h a = frag_a_vec(Arow, 512, kk * 32, lane);
    v16h b = frag_b_sw(Bm, 32, nt, kk, lane);
    acc = wmma16(a, b, acc);
  }
  int n = nt * 16 + (lane & 15);
  int mh = (lane >> 4) * 8;
#pragma unroll
  for (int v = 0; v < 8; ++v) {
    int row = mt * 16 + v + mh;
    if (which == 0) pre_h[row * 512 + n] = (f16)tanhf(acc[v] + b_ae[n]);
    else            encJ[row * 512 + n] = acc[v] + b_le[n];
  }
}

// ---------------- persistent recurrent decoder (1 block, 512 threads) --------
__global__ __launch_bounds__(512) void k_decoder(
    const f16* __restrict__ pre_h, const f16* __restrict__ hs_h,
    const f16* __restrict__ eys_h, const f16* __restrict__ WadSW,
    const f16* __restrict__ W0, const f16* __restrict__ W1,
    const float* __restrict__ b_ad,
    const float* __restrict__ b_ih0, const float* __restrict__ b_hh0,
    const float* __restrict__ b_ih1, const float* __restrict__ b_hh1,
    const int* __restrict__ hlens, f16* __restrict__ hdec_h) {
  __shared__ __align__(16) f16 ah[8 * 1536];  // rows: [ey | att_c | z0]
  __shared__ __align__(16) f16 z1h[8 * 512];
  __shared__ __align__(16) f16 qh[8 * 512];
  __shared__ float ew[8 * 256];
  __shared__ int hl[8];

  int tid = threadIdx.x;
  int lane = tid & 31, wv = tid >> 5;  // 16 waves
  int m = lane & 15, kb = (lane >> 4) * 8;

  for (int i = tid; i < 8 * 1536; i += 512) ah[i] = (f16)0.0f;
  for (int i = tid; i < 8 * 512; i += 512) z1h[i] = (f16)0.0f;
  if (tid < 8) hl[tid] = hlens[tid];
  float c0r[2][8], c1r[2][8];
#pragma unroll
  for (int i = 0; i < 2; ++i)
#pragma unroll
    for (int v = 0; v < 8; ++v) { c0r[i][v] = 0.f; c1r[i][v] = 0.f; }
  __syncthreads();

  v8f zero = {};
  v16h zh = {};
  for (int u = 0; u < U_; ++u) {
    // -- q = tanh(z0 @ Wad + b_ad); z0 in ah cols [1024,1536)
#pragma unroll
    for (int i = 0; i < 2; ++i) {
      int nt = wv * 2 + i;
      v8f acc = zero;
      for (int kk = 0; kk < 16; ++kk) {
        v16h a = zh;
        if (m < 8) {
          const f16* p = ah + m * 1536 + 1024 + kk * 32 + kb;
          a = combine16(*(const v8h*)p, *(const v8h*)(p + 16));
        }
        acc = wmma16(a, frag_b_sw(WadSW, 32, nt, kk, lane), acc);
      }
      if (lane < 16) {
        int n = nt * 16 + lane;
#pragma unroll
        for (int v = 0; v < 8; ++v) qh[v * 512 + n] = (f16)tanhf(acc[v] + b_ad[n]);
      }
    }
    __syncthreads();

    // -- e[b][t] = pre_enc[b,t,:]·q[b,:] (masked), vectorized f16 loads
    for (int r = 0; r < 4; ++r) {
      int idx = r * 512 + tid;
      int bb = idx >> 8, t = idx & 255;
      const f16* pe = pre_h + (bb * T_ + t) * 512;
      const f16* qq = qh + bb * 512;
      float s = 0.f;
      for (int k = 0; k < 512; k += 8) {
        v8h pa = *(const v8h*)(pe + k);
        v8h qa = *(const v8h*)(qq + k);
#pragma unroll
        for (int j = 0; j < 8; ++j) s += (float)pa[j] * (float)qa[j];
      }
      ew[idx] = (t < hl[bb]) ? s : -1e9f;
    }
    __syncthreads();
    if (tid < 8) {  // softmax over T per batch row
      float mx = -1e30f;
      for (int t = 0; t < 256; ++t) mx = fmaxf(mx, ew[tid * 256 + t]);
      float sm = 0.f;
      for (int t = 0; t < 256; ++t) { float v = expf(ew[tid * 256 + t] - mx); ew[tid * 256 + t] = v; sm += v; }
      float inv = 1.f / sm;
      for (int t = 0; t < 256; ++t) ew[tid * 256 + t] *= inv;
    }
    __syncthreads();
    {  // att_c = w·hs ; fill ey
      int bb = tid >> 6, d0 = (tid & 63) * 8;
      float a8[8];
#pragma unroll
      for (int q = 0; q < 8; ++q) a8[q] = 0.f;
      for (int t = 0; t < 256; ++t) {
        float w = ew[bb * 256 + t];
        v8h h = *(const v8h*)(hs_h + (bb * T_ + t) * 512 + d0);
#pragma unroll
        for (int q = 0; q < 8; ++q) a8[q] += w * (float)h[q];
      }
#pragma unroll
      for (int q = 0; q < 8; ++q) ah[bb * 1536 + 512 + d0 + q] = (f16)a8[q];
      v8h ey = *(const v8h*)(eys_h + (bb * U_ + u) * 512 + d0);
      *(v8h*)(ah + bb * 1536 + d0) = ey;
    }
    __syncthreads();

    // -- LSTM0: [x|z0] (16x1536) @ W0 (1536x2048); wave owns 2 j-tiles x 4 gates
    v8f acc[2][4];
#pragma unroll
    for (int i = 0; i < 2; ++i) {
      int jt = wv * 2 + i;
#pragma unroll
      for (int g = 0; g < 4; ++g) acc[i][g] = zero;
      for (int kk = 0; kk < 48; ++kk) {
        v16h a = zh;
        if (m < 8) {
          const f16* p = ah + m * 1536 + kk * 32 + kb;
          a = combine16(*(const v8h*)p, *(const v8h*)(p + 16));
        }
#pragma unroll
        for (int g = 0; g < 4; ++g)
          acc[i][g] = wmma16(a, frag_b_sw(W0, 128, g * 32 + jt, kk, lane), acc[i][g]);
      }
    }
    __syncthreads();
    if (lane < 16) {
#pragma unroll
      for (int i = 0; i < 2; ++i) {
        int j = (wv * 2 + i) * 16 + lane;
#pragma unroll
        for (int v = 0; v < 8; ++v) {
          float gi = acc[i][0][v] + b_ih0[j] + b_hh0[j];
          float gf = acc[i][1][v] + b_ih0[512 + j] + b_hh0[512 + j];
          float gg = acc[i][2][v] + b_ih0[1024 + j] + b_hh0[1024 + j];
          float go = acc[i][3][v] + b_ih0[1536 + j] + b_hh0[1536 + j];
          float c = sigm(gf) * c0r[i][v] + sigm(gi) * tanhf(gg);
          c0r[i][v] = c;
          ah[v * 1536 + 1024 + j] = (f16)(sigm(go) * tanhf(c));
        }
      }
    }
    __syncthreads();

    // -- LSTM1: [z0_new|z1_old] (16x1024) @ W1 (1024x2048)
#pragma unroll
    for (int i = 0; i < 2; ++i) {
      int jt = wv * 2 + i;
#pragma unroll
      for (int g = 0; g < 4; ++g) acc[i][g] = zero;
      for (int kk = 0; kk < 32; ++kk) {
        v16h a = zh;
        if (m < 8) {
          const f16* p = (kk < 16) ? (ah + m * 1536 + 1024 + kk * 32 + kb)
                                   : (z1h + m * 512 + (kk - 16) * 32 + kb);
          a = combine16(*(const v8h*)p, *(const v8h*)(p + 16));
        }
#pragma unroll
        for (int g = 0; g < 4; ++g)
          acc[i][g] = wmma16(a, frag_b_sw(W1, 128, g * 32 + jt, kk, lane), acc[i][g]);
      }
    }
    __syncthreads();
    if (lane < 16) {
#pragma unroll
      for (int i = 0; i < 2; ++i) {
        int j = (wv * 2 + i) * 16 + lane;
#pragma unroll
        for (int v = 0; v < 8; ++v) {
          float gi = acc[i][0][v] + b_ih1[j] + b_hh1[j];
          float gf = acc[i][1][v] + b_ih1[512 + j] + b_hh1[512 + j];
          float gg = acc[i][2][v] + b_ih1[1024 + j] + b_hh1[1024 + j];
          float go = acc[i][3][v] + b_ih1[1536 + j] + b_hh1[1536 + j];
          float c = sigm(gf) * c1r[i][v] + sigm(gi) * tanhf(gg);
          c1r[i][v] = c;
          f16 h1 = (f16)(sigm(go) * tanhf(c));
          z1h[v * 512 + j] = h1;
          hdec_h[(v * U_ + u) * 512 + j] = h1;
        }
      }
    }
    __syncthreads();
  }
}

// ---------------- decJ = h_dec @ W_lin_dec (no bias) ----------------
__global__ __launch_bounds__(256) void k_decj(const f16* __restrict__ hdec_h,
                                              const f16* __restrict__ WldSW,
                                              float* __restrict__ decJ) {
  int lane = threadIdx.x & 31, wv = threadIdx.x >> 5;
  int gid = blockIdx.x * 8 + wv;  // 1024 tiles = 32mt x 32nt
  int mt = gid >> 5, nt = gid & 31;
  const f16* Arow = hdec_h + (mt * 16) * 512;
  v8f acc = {};
  for (int kk = 0; kk < 16; ++kk) {
    v16h a = frag_a_vec(Arow, 512, kk * 32, lane);
    acc = wmma16(a, frag_b_sw(WldSW, 32, nt, kk, lane), acc);
  }
  int n = nt * 16 + (lane & 15);
  int mh = (lane >> 4) * 8;
#pragma unroll
  for (int v = 0; v < 8; ++v) decJ[(mt * 16 + v + mh) * 512 + n] = acc[v];
}

// ------- joint: tanh(encJ+decJ) @ W_out + b_out (dominant, 67 GFLOP) ---------
// Block = (b, 16 t's, 8 u's): M=128.  A tiles live in registers (tanh fused);
// W_out K-strips double-buffered into LDS via async-to-LDS when available.
__global__ __launch_bounds__(256) void k_joint(const float* __restrict__ encJ,
                                               const float* __restrict__ decJ,
                                               const f16* __restrict__ WoutSW,
                                               const float* __restrict__ b_out,
                                               float* __restrict__ out) {
  __shared__ __align__(16) f16 Bs[2][16 * 512];  // 2 x 16KB strips
  int tid = threadIdx.x, lane = tid & 31, wv = tid >> 5;
  int id = blockIdx.x;  // 1024 = b(8) * tt(16) * ug(8)
  int ug = id & 7;
  int tt = (id >> 3) & 15;
  int bb = id >> 7;
  int t0 = tt * 16;
  int u = ug * 8 + wv;
  int m = lane & 15, kb = (lane >> 4) * 8;
  const float* dj = decJ + (size_t)(bb * U_ + u) * 512;
  const float* ej = encJ + (size_t)(bb * T_ + t0 + m) * 512;

  // build this wave's A fragments in registers (fused tanh)
  v16h areg[16];
  for (int kk = 0; kk < 16; ++kk) {
#pragma unroll
    for (int e = 0; e < 16; ++e) {
      int k = kk * 32 + kb + (e < 8 ? e : e + 8);
      areg[kk][e] = (f16)tanhf(ej[k] + dj[k]);
    }
  }

  const uint4* wsrc = (const uint4*)WoutSW;  // 8 f16 per uint4
  int mh = (lane >> 4) * 8;

#if HAS_ASYNC_LDS
  // --- async double-buffered pipeline: ASYNCcnt-tracked LDS fill ---
  for (int i = tid; i < 1024; i += 256) {  // prologue: strip 0 -> Bs[0]
    int kk = i >> 6, off = i & 63;
    __builtin_amdgcn_global_load_async_to_lds_b128(
        (gv4i_p)(wsrc + (size_t)(kk * 32) * 64 + off),
        (lv4i_p)((uint4*)Bs[0] + kk * 64 + off), 0, 0);
  }
  for (int nt = 0; nt < 32; ++nt) {
    int p = nt & 1;
    __builtin_amdgcn_s_wait_asynccnt(0);  // my strip-nt portion landed
    __syncthreads();                      // all portions landed; buf[1-p] free
    if (nt + 1 < 32) {
      for (int i = tid; i < 1024; i += 256) {
        int kk = i >> 6, off = i & 63;
        __builtin_amdgcn_global_load_async_to_lds_b128(
            (gv4i_p)(wsrc + (size_t)(kk * 32 + nt + 1) * 64 + off),
            (lv4i_p)((uint4*)Bs[1 - p] + kk * 64 + off), 0, 0);
      }
    }
    v8f acc = {};
#pragma unroll
    for (int kk = 0; kk < 16; ++kk) {
      v16h b = *(const v16h*)(Bs[p] + kk * 512 + (lane << 4));
      acc = wmma16(areg[kk], b, acc);
    }
    int n = nt * 16 + m;
    if (n < OD_) {
      float bo = b_out[n];
#pragma unroll
      for (int v = 0; v < 8; ++v) {
        int t = t0 + v + mh;
        out[(((size_t)bb * T_ + t) * U_ + u) * OD_ + n] = acc[v] + bo;
      }
    }
  }
#else
  // --- synchronous fallback: VGPR round-trip staging ---
  uint4* bs4 = (uint4*)Bs[0];
  for (int nt = 0; nt < 32; ++nt) {
    __syncthreads();
    for (int i = tid; i < 1024; i += 256) {
      int kk = i >> 6, off = i & 63;
      bs4[kk * 64 + off] = wsrc[(size_t)(kk * 32 + nt) * 64 + off];
    }
    __syncthreads();
    v8f acc = {};
#pragma unroll
    for (int kk = 0; kk < 16; ++kk) {
      v16h b = *(const v16h*)(Bs[0] + kk * 512 + (lane << 4));
      acc = wmma16(areg[kk], b, acc);
    }
    int n = nt * 16 + m;
    if (n < OD_) {
      float bo = b_out[n];
#pragma unroll
      for (int v = 0; v < 8; ++v) {
        int t = t0 + v + mh;
        out[(((size_t)bb * T_ + t) * U_ + u) * OD_ + n] = acc[v] + bo;
      }
    }
  }
#endif
}

extern "C" void kernel_launch(void* const* d_in, const int* in_sizes, int n_in,
                              void* d_out, int out_size, void* d_ws, size_t ws_size,
                              hipStream_t stream) {
  (void)in_sizes; (void)n_in; (void)out_size; (void)ws_size;
  const float* hs    = (const float*)d_in[0];
  const int*   ys    = (const int*)d_in[1];
  const int*   hlens = (const int*)d_in[2];
  const float* emb   = (const float*)d_in[3];
  const float* W_ih0 = (const float*)d_in[4];
  const float* W_hh0 = (const float*)d_in[5];
  const float* b_ih0 = (const float*)d_in[6];
  const float* b_hh0 = (const float*)d_in[7];
  const float* W_ih1 = (const float*)d_in[8];
  const float* W_hh1 = (const float*)d_in[9];
  const float* b_ih1 = (const float*)d_in[10];
  const float* b_hh1 = (const float*)d_in[11];
  const float* W_ae  = (const float*)d_in[12];
  const float* b_ae  = (const float*)d_in[13];
  const float* W_ad  = (const float*)d_in[14];
  const float* b_ad  = (const float*)d_in[15];
  const float* W_le  = (const float*)d_in[16];
  const float* b_le  = (const float*)d_in[17];
  const float* W_ld  = (const float*)d_in[18];
  const float* W_out = (const float*)d_in[19];
  const float* b_out = (const float*)d_in[20];
  float* out = (float*)d_out;

  char* ws = (char*)d_ws;
  f16*   HS_H   = (f16*)(ws + OFF_HS_H);
  f16*   PRE_H  = (f16*)(ws + OFF_PRE_H);
  float* ENCJ   = (float*)(ws + OFF_ENCJ);
  float* DECJ   = (float*)(ws + OFF_DECJ);
  f16*   HDEC_H = (f16*)(ws + OFF_HDEC_H);
  f16*   EYS_H  = (f16*)(ws + OFF_EYS_H);
  f16*   WAE_H  = (f16*)(ws + OFF_WAE_H);
  f16*   WLE_H  = (f16*)(ws + OFF_WLE_H);
  f16*   WLD_H  = (f16*)(ws + OFF_WLD_H);
  f16*   WAD_H  = (f16*)(ws + OFF_WAD_H);
  f16*   WOUT_H = (f16*)(ws + OFF_WOUT_H);
  f16*   W0_H   = (f16*)(ws + OFF_W0_H);
  f16*   W1_H   = (f16*)(ws + OFF_W1_H);

  // prep: precision conversion + fragment swizzle
  k_cast<<<4096, 256, 0, stream>>>(HS_H, hs, B_ * T_ * E_);
  k_sw_direct<<<1024, 256, 0, stream>>>(WAE_H, W_ae, 32, 512 * 512);
  k_sw_direct<<<1024, 256, 0, stream>>>(WLE_H, W_le, 32, 512 * 512);
  k_sw_direct<<<1024, 256, 0, stream>>>(WLD_H, W_ld, 32, 512 * 512);
  k_sw_direct<<<1024, 256, 0, stream>>>(WAD_H, W_ad, 32, 512 * 512);
  k_sw_wout<<<1024, 256, 0, stream>>>(WOUT_H, W_out);
  k_sw_stack0<<<12288, 256, 0, stream>>>(W0_H, W_ih0, W_hh0);
  k_sw_stack1<<<8192, 256, 0, stream>>>(W1_H, W_ih1, W_hh1);
  k_emb<<<1024, 256, 0, stream>>>(EYS_H, emb, ys);

  // encoder projections (WMMA)
  k_enc<<<1024, 256, 0, stream>>>(HS_H, WAE_H, WLE_H, b_ae, b_le, PRE_H, ENCJ);

  // persistent attention + 2-layer LSTM recurrence (WMMA)
  k_decoder<<<1, 512, 0, stream>>>(PRE_H, HS_H, EYS_H, WAD_H, W0_H, W1_H, b_ad,
                                   b_ih0, b_hh0, b_ih1, b_hh1, hlens, HDEC_H);

  // decoder joint projection (WMMA)
  k_decj<<<128, 256, 0, stream>>>(HDEC_H, WLD_H, DECJ);

  // fused joint GEMM (dominant; WMMA + async LDS staging)
  k_joint<<<1024, 256, 0, stream>>>(ENCJ, DECJ, WOUT_H, b_out, out);
}
